// WarpedContextNet_74998718923216
// MI455X (gfx1250) — matrix operand order repaired and checked by reading, not compiled, not run on hardware
//
#include <hip/hip_runtime.h>
#include <hip/hip_bf16.h>
#include <math.h>

typedef __attribute__((ext_vector_type(16))) _Float16 v16h;
typedef __attribute__((ext_vector_type(8)))  float    v8f;
typedef unsigned int u32x4 __attribute__((ext_vector_type(4)));
typedef int          i32x4 __attribute__((ext_vector_type(4)));
typedef int          i32x8 __attribute__((ext_vector_type(8)));

#define NB 2
#define FH 512
#define FW 896
#define HH 256
#define WW 448
#define PP (HH*WW)        // 114688
#define NP (NB*PP)        // 229376

// ---------------------------------------------------------------------------
// Elementwise helpers (32-bit indexing, constant divisors)
// ---------------------------------------------------------------------------
__global__ void zero_fill(float* p, int n) {
    int i = blockIdx.x * blockDim.x + threadIdx.x;
    if (i < n) p[i] = 0.f;
}

// jax.image.resize 'bilinear' (antialiased) exact-2x downsample ==
// separable 4-tap [1,3,3,1]/8 with edge clamping.
__global__ void downsample_half(const float* __restrict__ src, float* __restrict__ dst,
                                int NC, float scale) {
    const int OH = FH >> 1, OW = FW >> 1;
    int i = blockIdx.x * blockDim.x + threadIdx.x;
    if (i >= NC * OH * OW) return;
    int ox = i % OW; int t = i / OW;
    int oy = t % OH; int pc = t / OH;
    const float wt[4] = {0.125f, 0.375f, 0.375f, 0.125f};
    const float* sp = src + (long)pc * FH * FW;
    float acc = 0.f;
    #pragma unroll
    for (int ty = 0; ty < 4; ++ty) {
        int yy = 2*oy - 1 + ty; yy = yy < 0 ? 0 : (yy >= FH ? FH-1 : yy);
        float wy = wt[ty];
        #pragma unroll
        for (int tx = 0; tx < 4; ++tx) {
            int xx = 2*ox - 1 + tx; xx = xx < 0 ? 0 : (xx >= FW ? FW-1 : xx);
            acc += wy * wt[tx] * sp[yy*FW + xx];
        }
    }
    dst[i] = acc * scale;
}

// ---------------------------------------------------------------------------
// TDM: DMA a linear block of `nel` f32 elements from global into LDS.
// D# per CDNA5 ISA ch.8: group0 {count=1, lds_addr, global_addr, type=2},
// group1 {data_size=4B, tensor_dim0=nel, tensor_dim1=1, tile_dim0=nel,
//         tile_dim1=1, dim0_stride=nel}. Groups 2/3 zero (dims 2..4 unused).
// ---------------------------------------------------------------------------
__device__ inline void tdm_load_f32(const float* gsrc, float* ldst, unsigned nel) {
    unsigned long long ga = (unsigned long long)(const void*)gsrc;
    unsigned ldsOff = (unsigned)(unsigned long long)(void*)ldst; // generic->LDS: low 32 bits
    u32x4 g0;
    g0[0] = 1u;                                                  // count=1
    g0[1] = ldsOff;                                              // lds_addr
    g0[2] = (unsigned)(ga & 0xFFFFFFFFu);                        // global_addr[31:0]
    g0[3] = (unsigned)((ga >> 32) & 0x01FFFFFFu) | (2u << 30);   // addr[56:32], type=2
    i32x8 g1;
    g1[0] = (int)(2u << 16);                                     // data_size=2 (4 bytes)
    g1[1] = (int)((nel & 0xFFFFu) << 16);                        // tensor_dim0[15:0]
    g1[2] = (int)((nel >> 16) | (1u << 16));                     // tensor_dim0[31:16], dim1=1
    g1[3] = (int)(nel << 16);                                    // tile_dim0 (=nel, <2^16)
    g1[4] = 1;                                                   // tile_dim1=1
    g1[5] = (int)nel;                                            // tensor_dim0_stride
    g1[6] = 0; g1[7] = 0;
    i32x4 gz = {0,0,0,0};
    i32x8 gz8 = {0,0,0,0,0,0,0,0};
#if __has_include(<hip/amd_detail/amd_gfx1250_TDM.h>)
    __builtin_amdgcn_tensor_load_to_lds(g0, g1, gz, gz, gz8, 0); // clang-23 form
    (void)gz8;
#else
    (void)gz8;
    __builtin_amdgcn_tensor_load_to_lds(g0, g1, gz, gz, 0);      // ROCm 7.2 form
#endif
}

// ---------------------------------------------------------------------------
// WMMA conv: D(16 pixels x 16 outch) += A(16 x 32 K) * B(32 K x 16 outch)
// A/B/D packed per CDNA5 ISA 7.12.2 wave32 layouts. f16 operands, f32 acc.
// Weights staged into LDS once per block via the Tensor Data Mover.
// Output geometry fixed at HH x WW (all layers), so tile decomposition uses
// compile-time divisors only.
// ---------------------------------------------------------------------------
template<int KH, int KW, int S, int PADY, int PADX, int CIN, int COUT>
__global__ __launch_bounds__(128)
void conv_wmma(const float* __restrict__ in, int IH, int IW,
               const float* __restrict__ wgt, const float* __restrict__ bias,
               const float* __restrict__ alpha, int useAct,
               float* __restrict__ out) {
    constexpr int K      = CIN * KH * KW;
    constexpr int NCH    = (K + 31) / 32;
    constexpr int CTIL   = (COUT + 15) / 16;
    constexpr int TILESX = WW / 16;
    constexpr int TOT    = NB * CTIL * HH * TILESX;
    constexpr bool STAGE = ((long)COUT * K * 4) <= 200*1024;

    __shared__ float lw[STAGE ? (COUT * K) : 1];
    if constexpr (STAGE) {
        if (threadIdx.x < 32) {                     // wave 0 issues the DMA
            tdm_load_f32(wgt, lw, (unsigned)(COUT * K));
            __builtin_amdgcn_s_wait_tensorcnt(0);   // s_wait_tensorcnt 0
        }
        __syncthreads();
    }

    const int lane = threadIdx.x & 31;
    int t = blockIdx.x * 4 + (threadIdx.x >> 5);
    if (t >= TOT) return;                           // wave-uniform exit
    const int xt = t % TILESX; int t2 = t / TILESX;
    const int y  = t2 % HH;    int t3 = t2 / HH;
    const int ct = t3 % CTIL;
    const int n  = t3 / CTIL;
    const int x0  = xt << 4;
    const int M   = lane & 15;        // A row (pixel within tile) / B,D column
    const int hh  = lane >> 4;        // lane half
    const int px  = x0 + M;           // output x for this lane's A row
    const int co  = (ct << 4) + M;    // out channel for this lane's B column
    const float* inN = in + (long)n * CIN * IH * IW;
    __builtin_prefetch(inN, 0, 1);    // global_prefetch_b8

    v8f acc = {};
    for (int ch = 0; ch < NCH; ++ch) {
        v16h A = {}; v16h B = {};
        #pragma unroll
        for (int v = 0; v < 8; ++v) {
            #pragma unroll
            for (int e = 0; e < 2; ++e) {
                // A layout: v<4 -> K = 2v+e+8*hh ; v>=4 -> K = 16+2(v-4)+e+8*hh
                int ka = ch*32 + ((v < 4) ? (2*v + e + 8*hh) : (16 + 2*(v-4) + e + 8*hh));
                float va = 0.f;
                if (ka < K) {
                    int ci = ka / (KH*KW);
                    int r  = ka % (KH*KW);
                    int iy = y  * S - PADY + (r / KW);
                    int ix = px * S - PADX + (r % KW);
                    if (iy >= 0 && iy < IH && ix >= 0 && ix < IW)
                        va = inN[ci*IH*IW + iy*IW + ix];
                }
                A[2*v+e] = (_Float16)va;
                // B layout: v<4 -> K = 2v+e+8*hh ; v>=4 -> K = 16+2(v-4)+e+8*hh
                int kb = ch*32 + ((v < 4) ? (2*v + e + 8*hh) : (16 + 2*(v-4) + e + 8*hh));
                float vb = 0.f;
                if (kb < K) {
                    if constexpr (STAGE) vb = lw[co*K + kb];      // ds_load
                    else                 vb = wgt[co*K + kb];     // OIHW flat
                }
                B[2*v+e] = (_Float16)vb;
            }
        }
        acc = __builtin_amdgcn_wmma_f32_16x16x32_f16(false, A, false, B,
                                                     (short)0, acc, false, false);
    }
    // D layout: VGPR d, lane -> M = d + 8*hh (pixel), N = lane&15 (out channel)
    const float b = bias[co];
    const float a = useAct ? alpha[co] : 0.f;
    float* outN = out + ((long)(n*COUT + co))*PP + y*WW;
    #pragma unroll
    for (int d = 0; d < 8; ++d) {
        float v = acc[d] + b;
        if (useAct && v < 0.f) v = a * v;
        outN[x0 + d + 8*hh] = v;
    }
}

// masknet conv3: 16 -> 1, 3x3 pad 1, no activation (direct)
__global__ void conv3x3_16to1(const float* __restrict__ in, const float* __restrict__ w,
                              const float* __restrict__ b, float* __restrict__ out) {
    int i = blockIdx.x * blockDim.x + threadIdx.x;
    if (i >= NB * PP) return;
    int x = i % WW; int t = i / WW;
    int y = t % HH; int n = t / HH;
    const float* inN = in + (long)n * 16 * PP;
    float acc = b[0];
    for (int ci = 0; ci < 16; ++ci) {
        #pragma unroll
        for (int ky = 0; ky < 3; ++ky) {
            int iy = y - 1 + ky;
            if (iy < 0 || iy >= HH) continue;
            #pragma unroll
            for (int kx = 0; kx < 3; ++kx) {
                int ix = x - 1 + kx;
                if (ix < 0 || ix >= WW) continue;
                acc += inN[ci*PP + iy*WW + ix] * w[(ci*3 + ky)*3 + kx];
            }
        }
    }
    out[i] = acc;
}

// ---------------------------------------------------------------------------
// Forward splat (fwarp): scatter-add with gaussian weights (f32 atomics).
// img: [N,2,h,w]; flo = s*img where s = t or (1-t) (t from tc plane ti).
// ---------------------------------------------------------------------------
__global__ void fwarp_splat(const float* __restrict__ img, const float* __restrict__ tc,
                            int ti, int oneMinusT,
                            float* __restrict__ imgw, float* __restrict__ onew) {
    int i = blockIdx.x * blockDim.x + threadIdx.x;
    if (i >= NB * 2 * PP) return;
    int pix = i % PP; int t0 = i / PP;
    int c = t0 % 2; int n = t0 / 2;
    int jj = pix % WW, ii = pix / WW;
    float tv = tc[(n*3 + ti)*PP + pix];
    float s  = oneMinusT ? (1.f - tv) : tv;
    float yf = s * img[(n*2 + 0)*PP + pix];   // "y" = flo[:,0]
    float xf = s * img[(n*2 + 1)*PP + pix];   // "x" = flo[:,1]
    float val = img[(n*2 + c)*PP + pix];
    float x1 = floorf(xf), y1 = floorf(yf);
    #pragma unroll
    for (int sxi = 0; sxi < 2; ++sxi) {
        #pragma unroll
        for (int syi = 0; syi < 2; ++syi) {
            float sx = x1 + (float)sxi, sy = y1 + (float)syi;
            float dx = xf - sx, dy = yf - sy;
            float wgt = expf(-(dx*dx + dy*dy));
            int r  = (int)sx + ii;
            int cc = (int)sy + jj;
            if (r >= 0 && r < HH && cc >= 0 && cc < WW) {
                int id = ((n*2 + c)*HH + r)*WW + cc;
                atomicAdd(&imgw[id], val * wgt);   // global_atomic_add_f32
                atomicAdd(&onew[id], wgt);
            }
        }
    }
}

__global__ void cfr_combine(const float* __restrict__ f01w, const float* __restrict__ n0,
                            const float* __restrict__ f10w, const float* __restrict__ n1,
                            const float* __restrict__ tc, int ti,
                            float* __restrict__ ft0, float* __restrict__ ft1) {
    int i = blockIdx.x * blockDim.x + threadIdx.x;
    if (i >= NB * 2 * PP) return;
    int pix = i % PP; int n = i / (2*PP);
    float t = tc[(n*3 + ti)*PP + pix];
    float a = f01w[i], b = f10w[i], na = n0[i], nb = n1[i];
    float v0 = -(1.f - t)*t*a + t*t*b;
    float v1 = (1.f - t)*(1.f - t)*a - t*(1.f - t)*b;
    float norm = (1.f - t)*na + t*nb;
    float m = (norm > 0.f) ? 1.f : 0.f;
    ft0[i] = (1.f - m)*v0 + m*(v0 / (norm + (1.f - m)));
    ft1[i] = (1.f - m)*v1 + m*(v1 / (norm + (1.f - m)));
}

__global__ void concat36(const float* __restrict__ xsA, const float* __restrict__ xsB,
                         const float* __restrict__ fA, const float* __restrict__ fB,
                         float* __restrict__ dst) {
    int i = blockIdx.x * blockDim.x + threadIdx.x;
    if (i >= NB * 36 * PP) return;
    int pix = i % PP; int t = i / PP;
    int ch = t % 36; int n = t / 36;
    float v;
    if (ch < 16)      v = xsA[(n*16 + ch)*PP + pix];
    else if (ch < 32) v = xsB[(n*16 + ch - 16)*PP + pix];
    else if (ch < 34) v = fA[(n*2 + ch - 32)*PP + pix];
    else              v = fB[(n*2 + ch - 34)*PP + pix];
    dst[i] = v;
}

// ---------------------------------------------------------------------------
// Fused backward warp (grid_sample bilinear, border pad + zeros-pad mask),
// sigmoid occlusion blend, and wm-weighted write into the 32-ch final concat.
// ---------------------------------------------------------------------------
struct BW { int x0, x1, y0, y1; float w00, w01, w10, w11, mask; };

__device__ inline BW bw_setup(float fx, float fy, int j, int i) {
    float gx = (float)j + fx, gy = (float)i + fy;
    float nx = 2.f*gx/(float)(WW - 1) - 1.f;
    float ny = 2.f*gy/(float)(HH - 1) - 1.f;
    float px = ((nx + 1.f)*(float)WW - 1.f)*0.5f;
    float py = ((ny + 1.f)*(float)HH - 1.f)*0.5f;
    float cx = fminf(fmaxf(px, 0.f), (float)(WW - 1));
    float cy = fminf(fmaxf(py, 0.f), (float)(HH - 1));
    float x0f = floorf(cx), y0f = floorf(cy);
    float wx = cx - x0f, wy = cy - y0f;
    BW r;
    r.x0 = min(max((int)x0f, 0), WW - 1);
    r.y0 = min(max((int)y0f, 0), HH - 1);
    r.x1 = min(r.x0 + 1, WW - 1);
    r.y1 = min(r.y0 + 1, HH - 1);
    r.w00 = (1.f - wx)*(1.f - wy); r.w01 = wx*(1.f - wy);
    r.w10 = (1.f - wx)*wy;         r.w11 = wx*wy;
    float fx0 = floorf(px), fy0 = floorf(py);
    float ux = px - fx0, uy = py - fy0;
    auto inb = [](float yy, float xx) {
        return (xx >= 0.f && xx < (float)WW && yy >= 0.f && yy < (float)HH) ? 1.f : 0.f;
    };
    float mv = (1.f - ux)*(1.f - uy)*inb(fy0, fx0)      + ux*(1.f - uy)*inb(fy0, fx0 + 1.f)
             + (1.f - ux)*uy*inb(fy0 + 1.f, fx0)        + ux*uy*inb(fy0 + 1.f, fx0 + 1.f);
    r.mask = (mv >= 0.999f) ? 1.f : 0.f;
    return r;
}

__device__ inline float bw_sample(const float* p, const BW& b) {
    return (b.w00*p[b.y0*WW + b.x0] + b.w01*p[b.y0*WW + b.x1] +
            b.w10*p[b.y1*WW + b.x0] + b.w11*p[b.y1*WW + b.x1]) * b.mask;
}

__global__ void bwarp_blend(const float* __restrict__ xsA, const float* __restrict__ xsB,
                            const float* __restrict__ fA,  const float* __restrict__ fB,
                            const float* __restrict__ occ, const float* __restrict__ tc,
                            int ti, int invW, int chOff, float* __restrict__ outcat) {
    int i = blockIdx.x * blockDim.x + threadIdx.x;
    if (i >= NB * PP) return;
    int pix = i % PP; int n = i / PP;
    int jj = pix % WW, ii = pix / WW;
    float t   = tc[(n*3 + ti)*PP + pix];
    float wmv = tc[(n*3 + 2)*PP + pix];
    if (invW) wmv = 1.f - wmv;
    float o0 = 1.f / (1.f + expf(-occ[n*PP + pix]));
    float o1 = 1.f - o0;
    float ca = (1.f - t)*o0, cb = t*o1;
    float den = ca + cb;
    BW bA = bw_setup(fA[(n*2 + 0)*PP + pix], fA[(n*2 + 1)*PP + pix], jj, ii);
    BW bB = bw_setup(fB[(n*2 + 0)*PP + pix], fB[(n*2 + 1)*PP + pix], jj, ii);
    #pragma unroll 4
    for (int c = 0; c < 16; ++c) {
        const float* pa = xsA + (n*16 + c)*PP;
        const float* pb = xsB + (n*16 + c)*PP;
        float F = (ca*bw_sample(pa, bA) + cb*bw_sample(pb, bB)) / den;
        outcat[((n*32 + chOff + c)*HH + ii)*WW + jj] = F * wmv;
    }
}

// ---------------------------------------------------------------------------
// Launch
// ---------------------------------------------------------------------------
extern "C" void kernel_launch(void* const* d_in, const int* in_sizes, int n_in,
                              void* d_out, int out_size, void* d_ws, size_t ws_size,
                              hipStream_t stream) {
    const float* x[3]  = {(const float*)d_in[0], (const float*)d_in[1], (const float*)d_in[2]};
    const float* flow[4] = {(const float*)d_in[3], (const float*)d_in[4],
                            (const float*)d_in[5], (const float*)d_in[6]};
    const float* enc   = (const float*)d_in[7];
    const float* c_w1  = (const float*)d_in[8];  const float* c_b1 = (const float*)d_in[9];
    const float* c_a1  = (const float*)d_in[10];
    const float* c_w2  = (const float*)d_in[11]; const float* c_b2 = (const float*)d_in[12];
    const float* c_a2  = (const float*)d_in[13];
    const float* m_w1  = (const float*)d_in[14]; const float* m_b1 = (const float*)d_in[15];
    const float* m_a1  = (const float*)d_in[16];
    const float* m_w2  = (const float*)d_in[17]; const float* m_b2 = (const float*)d_in[18];
    const float* m_a2  = (const float*)d_in[19];
    const float* m_w3  = (const float*)d_in[20]; const float* m_b3 = (const float*)d_in[21];
    const float* f_w   = (const float*)d_in[22]; const float* f_b  = (const float*)d_in[23];
    const float* f_a   = (const float*)d_in[24];
    float* out = (float*)d_out;

    // workspace carve-up (floats)
    float* ws = (float*)d_ws;
    size_t off = 0;
    float* tc    = ws + off; off += (size_t)3  * NP;   // [N,3,h,w] t1|t2|wm
    float* fl    = ws + off; off += (size_t)8  * NP;   // 4 flows x [N,2,h,w]
    float* tmp16 = ws + off; off += (size_t)16 * NP;   // convds stage-1 (reused)
    float* xs    = ws + off; off += (size_t)48 * NP;   // 3 x [N,16,h,w]
    float* facc  = ws + off; off += (size_t)8  * NP;   // f01w|n0|f10w|n1 (reused)
    float* ft    = ws + off; off += (size_t)8  * NP;   // ft0l|ft1l|ft1r|ft2r
    float* mcat  = ws + off; off += (size_t)36 * NP;   // masknet input (reused)
    float* m1    = ws + off; off += (size_t)32 * NP;
    float* m2    = ws + off; off += (size_t)16 * NP;
    float* occ   = ws + off; off += (size_t)2  * NP;   // occl | occr
    float* fcat  = ws + off; off += (size_t)32 * NP;   // final concat

    const int EB = 256;
    auto eb = [](int n){ return (n + 255) / 256; };
    const int b16 = (NB*1*HH*(WW/16)) / 4;   // 3584 blocks, COUT=16
    const int b32 = (NB*2*HH*(WW/16)) / 4;   // 7168 blocks, COUT=32

    // 1) half-res resizes
    downsample_half<<<eb(NB*3*PP), EB, 0, stream>>>(enc, tc, NB*3, 1.0f);
    for (int i = 0; i < 4; ++i)
        downsample_half<<<eb(NB*2*PP), EB, 0, stream>>>(flow[i], fl + (size_t)i*2*NP,
                                                        NB*2, 0.5f);
    // 2) convds per image (WMMA, TDM-staged weights)
    for (int i = 0; i < 3; ++i) {
        conv_wmma<3,3,2,1,1,3,16><<<b16, 128, 0, stream>>>(x[i], FH, FW, c_w1, c_b1, c_a1, 1,
                                                           tmp16);
        conv_wmma<3,3,1,1,1,16,16><<<b16, 128, 0, stream>>>(tmp16, HH, WW, c_w2, c_b2, c_a2, 1,
                                                            xs + (size_t)i*16*NP);
    }
    // 3) cfr left (t1 = tc plane 0) and right (t2 = tc plane 1)
    float* f01w = facc;                float* nn0 = facc + (size_t)2*NP;
    float* f10w = facc + (size_t)4*NP; float* nn1 = facc + (size_t)6*NP;
    for (int side = 0; side < 2; ++side) {
        zero_fill<<<eb(8*NP), EB, 0, stream>>>(facc, 8*NP);
        const float* fa = fl + (size_t)(2*side    )*2*NP;  // fl[0]/fl[2]
        const float* fb = fl + (size_t)(2*side + 1)*2*NP;  // fl[1]/fl[3]
        fwarp_splat<<<eb(NB*2*PP), EB, 0, stream>>>(fa, tc, side, 0, f01w, nn0);
        fwarp_splat<<<eb(NB*2*PP), EB, 0, stream>>>(fb, tc, side, 1, f10w, nn1);
        cfr_combine<<<eb(NB*2*PP), EB, 0, stream>>>(f01w, nn0, f10w, nn1, tc, side,
                                                    ft + (size_t)(2*side)*2*NP,
                                                    ft + (size_t)(2*side + 1)*2*NP);
    }
    // 4) masknet per side (WMMA for the two big convs)
    for (int side = 0; side < 2; ++side) {
        const float* xsA = xs + (size_t)(side    )*16*NP;
        const float* xsB = xs + (size_t)(side + 1)*16*NP;
        const float* ftA = ft + (size_t)(2*side    )*2*NP;
        const float* ftB = ft + (size_t)(2*side + 1)*2*NP;
        concat36<<<eb(NB*36*PP), EB, 0, stream>>>(xsA, xsB, ftA, ftB, mcat);
        conv_wmma<5,5,1,2,2,36,32><<<b32, 128, 0, stream>>>(mcat, HH, WW, m_w1, m_b1, m_a1, 1,
                                                            m1);
        conv_wmma<3,3,1,1,1,32,16><<<b16, 128, 0, stream>>>(m1, HH, WW, m_w2, m_b2, m_a2, 1,
                                                            m2);
        conv3x3_16to1<<<eb(NB*PP), EB, 0, stream>>>(m2, m_w3, m_b3, occ + (size_t)side*NP);
    }
    // 5) fused bwarp + sigmoid blend + wm weighting into final concat
    for (int side = 0; side < 2; ++side) {
        bwarp_blend<<<eb(NB*PP), EB, 0, stream>>>(
            xs + (size_t)(side    )*16*NP, xs + (size_t)(side + 1)*16*NP,
            ft + (size_t)(2*side  )*2*NP,  ft + (size_t)(2*side + 1)*2*NP,
            occ + (size_t)side*NP, tc, side, side, side*16, fcat);
    }
    // 6) final 1x1 conv 32->16 + PReLU (WMMA, TDM-staged weights) -> d_out
    conv_wmma<1,1,1,0,0,32,16><<<b16, 128, 0, stream>>>(fcat, HH, WW, f_w, f_b, f_a, 1, out);
}
